// DiscreteContinuousConvS2_28905129902156
// MI455X (gfx1250) — matrix operand order repaired and checked by reading, not compiled, not run on hardware
//
#include <hip/hip_runtime.h>

// ---- CDNA5 wave32 WMMA types --------------------------------------------
typedef __attribute__((ext_vector_type(16))) _Float16 v16h;
typedef __attribute__((ext_vector_type(8)))  _Float16 v8h;
typedef __attribute__((ext_vector_type(8)))  float    v8f;

// ---- problem constants (from reference) ---------------------------------
#define H_IN      128
#define H_OUT     128
#define LW        256   // L_IN == L_OUT
#define C_INN     32
#define C_OUTN    32
#define KKER      3
#define NNZ_ROW   32
#define KDIM      96    // C_IN * K  (== weight's flat [c,ik] order: r = c*3+ik)
#define YT_STRIDE 104   // padded LDS row (halves): 208B rows -> 16B-aligned frags

// One block per output latitude. 256 threads = 8 wave32s.
__global__ __launch_bounds__(256)
void disco_s2_fused_kernel(const float* __restrict__ x,         // [32,128,256]
                           const float* __restrict__ psi_vals,  // [12288]
                           const float* __restrict__ weight,    // [32,32,3] flat = [o][r]
                           const float* __restrict__ bias,      // [32]
                           const int*   __restrict__ psi_lat_in,// [12288]
                           const int*   __restrict__ psi_lon_in,// [12288]
                           float*       __restrict__ out)       // [32,128,256]
{
  __shared__ _Float16 yt[LW * YT_STRIDE];     // y transposed: [p][r], f16
  __shared__ _Float16 wh[C_OUTN * KDIM];      // f16 weights [o][r]
  __shared__ float    sval[KDIM];
  __shared__ int      soff[KDIM];             // lat_in * 256
  __shared__ int      slon[KDIM];
  __shared__ float    sbias[C_OUTN];

  const int t       = threadIdx.x;
  const int lat_out = blockIdx.x;
  const int lane    = t & 31;
  const int wave    = t >> 5;

  // ---------------- stage 0: stage metadata / weights into LDS -----------
  if (t < KDIM) {
    // structured COO: nz = ik*(H_OUT*32) + lat_out*32 + j
    const int ik = t >> 5, j = t & 31;
    const int nz = ik * (H_OUT * NNZ_ROW) + lat_out * NNZ_ROW + j;
    sval[t] = psi_vals[nz];
    soff[t] = psi_lat_in[nz] * LW;
    slon[t] = psi_lon_in[nz];
  }
  if (t < C_OUTN) sbias[t] = bias[t];
  for (int i = t; i < C_OUTN * KDIM; i += 256)
    wh[i] = (_Float16)weight[i];
  __syncthreads();

  // ---------------- stage A: gather-reduce y[r][p] (f32 acc, f16 store) --
  // wave owns rows r = wave + 8*rr; lane owns p = ch*32 + lane (8 chunks).
  for (int rr = 0; rr < 12; ++rr) {
    const int r  = wave + rr * 8;      // K index, r = c*3 + ik
    const int c  = r / 3;
    const int ik = r - c * 3;
    const float* __restrict__ xc = x + c * (H_IN * LW);

    float acc[8];
#pragma unroll
    for (int ch = 0; ch < 8; ++ch) acc[ch] = 0.0f;

    for (int j = 0; j < NNZ_ROW; ++j) {
      const int tn = ik * NNZ_ROW + j;      // wave-uniform LDS broadcast
      const float v  = sval[tn];
      const int   lo = slon[tn];
      const float* __restrict__ row = xc + soff[tn];
#pragma unroll
      for (int ch = 0; ch < 8; ++ch) {
        const int p = ch * 32 + lane;       // lanes -> consecutive lon: coalesced
        acc[ch] += v * row[(lo + p) & (LW - 1)];
      }
    }
#pragma unroll
    for (int ch = 0; ch < 8; ++ch) {
      const int p = ch * 32 + lane;
      yt[p * YT_STRIDE + r] = (_Float16)acc[ch];
    }
  }
  __syncthreads();

  // ---------------- stage B: out[o][p] = W[o][r] * y[r][p] via WMMA ------
  // v_wmma_f32_16x16x32_f16; M=32 (2 o-tiles), N=256 (16 p-tiles, 2/wave), K=96 (3 steps)
  const int n  = lane & 15;   // N (and A's M) index
  const int hi = lane >> 4;   // half-wave select

  // A fragments: 16x32, element h <-> k = ks*32 + 16*hi + h (contiguous halves)
  v16h afrag[2][3];
#pragma unroll
  for (int ot = 0; ot < 2; ++ot) {
#pragma unroll
    for (int ks = 0; ks < 3; ++ks) {
      v16h a;
#pragma unroll
      for (int h = 0; h < 16; ++h) {
        const int kk  = h + (hi << 4);
        a[h] = wh[(ot * 16 + n) * KDIM + ks * 32 + kk];
      }
      afrag[ot][ks] = a;
    }
  }

  v8f accw[2][2];
#pragma unroll
  for (int ot = 0; ot < 2; ++ot)
#pragma unroll
    for (int pti = 0; pti < 2; ++pti)
#pragma unroll
      for (int e = 0; e < 8; ++e) accw[ot][pti][e] = 0.0f;

#pragma unroll
  for (int ks = 0; ks < 3; ++ks) {
#pragma unroll
    for (int pti = 0; pti < 2; ++pti) {
      const int p = (wave * 2 + pti) * 16 + n;
      // B fragment: 32x16, element h <-> k = ks*32 + 16*hi + h (contiguous halves)
      const _Float16* bp = &yt[p * YT_STRIDE + ks * 32 + (hi << 4)];
      const v8h b0 = *(const v8h*)bp;          // 16B-aligned ds_load_b128
      const v8h b1 = *(const v8h*)(bp + 8);
      v16h b;
#pragma unroll
      for (int e = 0; e < 8; ++e) { b[e] = b0[e]; b[8 + e] = b1[e]; }

      accw[0][pti] = __builtin_amdgcn_wmma_f32_16x16x32_f16(
          false, afrag[0][ks], false, b, (short)0, accw[0][pti], false, false);
      accw[1][pti] = __builtin_amdgcn_wmma_f32_16x16x32_f16(
          false, afrag[1][ks], false, b, (short)0, accw[1][pti], false, false);
    }
  }

  // D layout: element i <-> m = i + 8*hi, col n; lanes 0-15 give 64B stores
#pragma unroll
  for (int ot = 0; ot < 2; ++ot) {
#pragma unroll
    for (int pti = 0; pti < 2; ++pti) {
      const int p = (wave * 2 + pti) * 16 + n;
#pragma unroll
      for (int i = 0; i < 8; ++i) {
        const int o = ot * 16 + i + (hi << 3);
        out[(o * H_OUT + lat_out) * LW + p] = accw[ot][pti][i] + sbias[o];
      }
    }
  }
}

extern "C" void kernel_launch(void* const* d_in, const int* in_sizes, int n_in,
                              void* d_out, int out_size, void* d_ws, size_t ws_size,
                              hipStream_t stream) {
  (void)in_sizes; (void)n_in; (void)d_ws; (void)ws_size; (void)out_size;
  const float* x          = (const float*)d_in[0];
  const float* psi_vals   = (const float*)d_in[1];
  const float* weight     = (const float*)d_in[2];
  const float* bias       = (const float*)d_in[3];
  // d_in[4] = psi_ik, d_in[5] = psi_lat_out: implied by structured nz ordering
  const int*   psi_lat_in = (const int*)d_in[6];
  const int*   psi_lon_in = (const int*)d_in[7];
  float* out = (float*)d_out;

  disco_s2_fused_kernel<<<H_OUT, 256, 0, stream>>>(
      x, psi_vals, weight, bias, psi_lat_in, psi_lon_in, out);
}
